// MLP_73074573574220
// MI455X (gfx1250) — matrix operand order repaired and checked by reading, not compiled, not run on hardware
//
#include <hip/hip_runtime.h>
#include <hip/hip_bf16.h>
#include <math.h>

#define NUM_EXPERTS 8
#define TOP_K 2
#define D_MODEL 1024
#define HIDDEN 1024
#define TWO_H 2048
#define N_TOKENS 2048

// Tiling
#define BM 128
#define BN 64
#define BK 32
#define LDT 40   // padded tile row length in bf16 elems (80 bytes, multiple of 16B)

typedef __attribute__((ext_vector_type(16))) __bf16 v16bf;
typedef __attribute__((ext_vector_type(8)))  __bf16 v8bf;
typedef __attribute__((ext_vector_type(4)))  __bf16 v4bf;
typedef __attribute__((ext_vector_type(8)))  float  v8f;

// ---------- bf16 conversion helpers (bit-exact, no reliance on __bf16 casts) ----------
static __device__ __forceinline__ __bf16 f2bf(float f) {
    union { float f; unsigned u; } uf; uf.f = f;
    unsigned r = uf.u + 0x7FFFu + ((uf.u >> 16) & 1u);   // round-to-nearest-even
    union { unsigned short s; __bf16 b; } ub; ub.s = (unsigned short)(r >> 16);
    return ub.b;
}
static __device__ __forceinline__ float bf2f(__bf16 b) {
    union { unsigned short s; __bf16 b; } ub; ub.b = b;
    union { unsigned u; float f; } uf; uf.u = ((unsigned)ub.s) << 16;
    return uf.f;
}

// ---------- WMMA wrapper ----------
static __device__ __forceinline__ v8f wmma_bf16(v16bf a, v16bf b, v8f c) {
    // (neg_a, A, neg_b, B, c_mod, C, reuse_a, reuse_b)
    return __builtin_amdgcn_wmma_f32_16x16x32_bf16(false, a, false, b, (short)0, c, false, false);
}

// A fragment (16x32, MxK): lane<16 holds K {0..7, 16..23}, lane>=16 holds {8..15, 24..31}
static __device__ __forceinline__ v16bf load_a_frag(const __bf16* tile, int rowBase, int l16, int lh) {
    const __bf16* p = tile + (size_t)(rowBase + l16) * LDT + lh * 8;
    v8bf x0 = *(const v8bf*)p;          // K = kbase + 0..7
    v8bf x1 = *(const v8bf*)(p + 16);   // K = 16 + kbase + 0..7
    return __builtin_shufflevector(x0, x1, 0,1,2,3,4,5,6,7,8,9,10,11,12,13,14,15);
}
// B fragment (32x16, KxN) from a transposed LDS tile Bt[n][k]:
// lane<16 holds K {0..7, 16..23}, lane>=16 holds {8..15, 24..31}
static __device__ __forceinline__ v16bf load_b_frag(const __bf16* tileT, int colBase, int l16, int lh) {
    const __bf16* p = tileT + (size_t)(colBase + l16) * LDT + lh * 8;
    v8bf x0 = *(const v8bf*)p;
    v8bf x1 = *(const v8bf*)(p + 16);
    return __builtin_shufflevector(x0, x1, 0,1,2,3,4,5,6,7,8,9,10,11,12,13,14,15);
}

static __device__ __forceinline__ void split4(float4 v, v4bf* hi, v4bf* lo) {
    __bf16 h0 = f2bf(v.x), h1 = f2bf(v.y), h2 = f2bf(v.z), h3 = f2bf(v.w);
    __bf16 l0 = f2bf(v.x - bf2f(h0)), l1 = f2bf(v.y - bf2f(h1));
    __bf16 l2 = f2bf(v.z - bf2f(h2)), l3 = f2bf(v.w - bf2f(h3));
    *hi = (v4bf){h0, h1, h2, h3};
    *lo = (v4bf){l0, l1, l2, l3};
}

// =====================================================================
// Gates kernel: G[n][e] = sum_k expert_p[n,k] * (expert_idxs[n,k]==e)
// =====================================================================
__global__ __launch_bounds__(256) void moe_gates_kernel(
    const float* __restrict__ p, const int* __restrict__ idx, float* __restrict__ G)
{
    int n = blockIdx.x * blockDim.x + threadIdx.x;
    if (n >= N_TOKENS) return;
    int   i0 = idx[n * TOP_K + 0], i1 = idx[n * TOP_K + 1];
    float p0 = p[n * TOP_K + 0],  p1 = p[n * TOP_K + 1];
#pragma unroll
    for (int e = 0; e < NUM_EXPERTS; ++e) {
        float s = ((i0 == e) ? p0 : 0.0f) + ((i1 == e) ? p1 : 0.0f);
        G[(size_t)n * NUM_EXPERTS + e] = s;
    }
}

// =====================================================================
// Up kernel (one expert): act[n, j] = gelu(h) * (g + 1),
//   h = (x @ w_up)[n, j], g = (x @ w_up)[n, j + HIDDEN]
// 3xBF16 split GEMM, both halves computed in the same wave tiles.
// =====================================================================
__global__ __launch_bounds__(256) void moe_up_kernel(
    const float* __restrict__ x, const float* __restrict__ w_up_e,
    float* __restrict__ act)
{
    __shared__ __bf16 sA_hi[BM * LDT];
    __shared__ __bf16 sA_lo[BM * LDT];
    __shared__ __bf16 sBhT_hi[BN * LDT];
    __shared__ __bf16 sBhT_lo[BN * LDT];
    __shared__ __bf16 sBgT_hi[BN * LDT];
    __shared__ __bf16 sBgT_lo[BN * LDT];

    const int tid  = threadIdx.x;
    const int lane = tid & 31;
    const int l16  = lane & 15, lh = lane >> 4;
    const int wave = tid >> 5;
    const int wm   = wave & 3;    // 4 row-groups of 32 rows
    const int wn   = wave >> 2;   // 2 col-groups of 32 cols
    const int rowBlk = blockIdx.y * BM;
    const int colBlk = blockIdx.x * BN;   // within h half [0, HIDDEN)

    v8f accH[2][2], accG[2][2];
#pragma unroll
    for (int i = 0; i < 2; ++i)
#pragma unroll
        for (int j = 0; j < 2; ++j) { accH[i][j] = (v8f)0.0f; accG[i][j] = (v8f)0.0f; }

    for (int kb = 0; kb < D_MODEL; kb += BK) {
        // ---- fill A tile: 128x32 f32 -> hi/lo bf16 ----
#pragma unroll
        for (int it = 0; it < (BM * BK / 4) / 256; ++it) {
            int s = tid + it * 256;
            int row = s >> 3, kv = s & 7;
            const float* src = x + (size_t)(rowBlk + row) * D_MODEL + kb + kv * 4;
            if (kb + BK < D_MODEL) __builtin_prefetch(src + BK, 0, 1);
            float4 v = *(const float4*)src;
            v4bf hi, lo; split4(v, &hi, &lo);
            *(v4bf*)&sA_hi[row * LDT + kv * 4] = hi;
            *(v4bf*)&sA_lo[row * LDT + kv * 4] = lo;
        }
        // ---- fill B tiles (h and g halves), transposed into LDS ----
#pragma unroll
        for (int it = 0; it < (BK * BN / 4) / 256; ++it) {
            int s = tid + it * 256;
            int kr = s >> 4, nv = s & 15;
            const float* src = w_up_e + (size_t)(kb + kr) * TWO_H + colBlk + nv * 4;
            if (kb + BK < D_MODEL) __builtin_prefetch(src + (size_t)BK * TWO_H, 0, 1);
            float4 vh = *(const float4*)src;
            float4 vg = *(const float4*)(src + HIDDEN);
            v4bf hih, loh, hig, log_;
            split4(vh, &hih, &loh);
            split4(vg, &hig, &log_);
#pragma unroll
            for (int j = 0; j < 4; ++j) {
                int n = nv * 4 + j;
                sBhT_hi[n * LDT + kr] = hih[j];
                sBhT_lo[n * LDT + kr] = loh[j];
                sBgT_hi[n * LDT + kr] = hig[j];
                sBgT_lo[n * LDT + kr] = log_[j];
            }
        }
        __syncthreads();

        v16bf aH[2], aL[2];
#pragma unroll
        for (int tm = 0; tm < 2; ++tm) {
            aH[tm] = load_a_frag(sA_hi, wm * 32 + tm * 16, l16, lh);
            aL[tm] = load_a_frag(sA_lo, wm * 32 + tm * 16, l16, lh);
        }
#pragma unroll
        for (int tn = 0; tn < 2; ++tn) {
            int cb = wn * 32 + tn * 16;
            v16bf bhH = load_b_frag(sBhT_hi, cb, l16, lh);
            v16bf bhL = load_b_frag(sBhT_lo, cb, l16, lh);
            v16bf bgH = load_b_frag(sBgT_hi, cb, l16, lh);
            v16bf bgL = load_b_frag(sBgT_lo, cb, l16, lh);
#pragma unroll
            for (int tm = 0; tm < 2; ++tm) {
                accH[tm][tn] = wmma_bf16(aL[tm], bhH, accH[tm][tn]);
                accH[tm][tn] = wmma_bf16(aH[tm], bhL, accH[tm][tn]);
                accH[tm][tn] = wmma_bf16(aH[tm], bhH, accH[tm][tn]);
                accG[tm][tn] = wmma_bf16(aL[tm], bgH, accG[tm][tn]);
                accG[tm][tn] = wmma_bf16(aH[tm], bgL, accG[tm][tn]);
                accG[tm][tn] = wmma_bf16(aH[tm], bgH, accG[tm][tn]);
            }
        }
        __syncthreads();
    }

    // ---- epilogue: exact GELU(h) * (g + 1) ----
#pragma unroll
    for (int tm = 0; tm < 2; ++tm)
#pragma unroll
        for (int tn = 0; tn < 2; ++tn)
#pragma unroll
            for (int r = 0; r < 8; ++r) {
                int row = rowBlk + wm * 32 + tm * 16 + lh * 8 + r;
                int col = colBlk + wn * 32 + tn * 16 + l16;
                float h = accH[tm][tn][r];
                float g = accG[tm][tn][r];
                float a = 0.5f * h * (1.0f + erff(h * 0.70710678118654752f)) * (g + 1.0f);
                act[(size_t)row * HIDDEN + col] = a;
            }
}

// =====================================================================
// Down kernel (one expert): y[n,:] (+)= gate[n,e] * (act @ w_down)[n,:]
// =====================================================================
__global__ __launch_bounds__(256) void moe_down_kernel(
    const float* __restrict__ act, const float* __restrict__ w_down_e,
    const float* __restrict__ gates, int expert, int beta,
    float* __restrict__ y)
{
    __shared__ __bf16 sA_hi[BM * LDT];
    __shared__ __bf16 sA_lo[BM * LDT];
    __shared__ __bf16 sBT_hi[BN * LDT];
    __shared__ __bf16 sBT_lo[BN * LDT];
    __shared__ float  sG[BM];

    const int tid  = threadIdx.x;
    const int lane = tid & 31;
    const int l16  = lane & 15, lh = lane >> 4;
    const int wave = tid >> 5;
    const int wm   = wave & 3;
    const int wn   = wave >> 2;
    const int rowBlk = blockIdx.y * BM;
    const int colBlk = blockIdx.x * BN;

    if (tid < BM)
        sG[tid] = gates[(size_t)(rowBlk + tid) * NUM_EXPERTS + expert];

    v8f acc[2][2];
#pragma unroll
    for (int i = 0; i < 2; ++i)
#pragma unroll
        for (int j = 0; j < 2; ++j) acc[i][j] = (v8f)0.0f;

    for (int kb = 0; kb < HIDDEN; kb += BK) {
#pragma unroll
        for (int it = 0; it < (BM * BK / 4) / 256; ++it) {
            int s = tid + it * 256;
            int row = s >> 3, kv = s & 7;
            const float* src = act + (size_t)(rowBlk + row) * HIDDEN + kb + kv * 4;
            if (kb + BK < HIDDEN) __builtin_prefetch(src + BK, 0, 1);
            float4 v = *(const float4*)src;
            v4bf hi, lo; split4(v, &hi, &lo);
            *(v4bf*)&sA_hi[row * LDT + kv * 4] = hi;
            *(v4bf*)&sA_lo[row * LDT + kv * 4] = lo;
        }
#pragma unroll
        for (int it = 0; it < (BK * BN / 4) / 256; ++it) {
            int s = tid + it * 256;
            int kr = s >> 4, nv = s & 15;
            const float* src = w_down_e + (size_t)(kb + kr) * D_MODEL + colBlk + nv * 4;
            if (kb + BK < HIDDEN) __builtin_prefetch(src + (size_t)BK * D_MODEL, 0, 1);
            float4 v = *(const float4*)src;
            v4bf hi, lo; split4(v, &hi, &lo);
#pragma unroll
            for (int j = 0; j < 4; ++j) {
                int n = nv * 4 + j;
                sBT_hi[n * LDT + kr] = hi[j];
                sBT_lo[n * LDT + kr] = lo[j];
            }
        }
        __syncthreads();

        v16bf aH[2], aL[2];
#pragma unroll
        for (int tm = 0; tm < 2; ++tm) {
            aH[tm] = load_a_frag(sA_hi, wm * 32 + tm * 16, l16, lh);
            aL[tm] = load_a_frag(sA_lo, wm * 32 + tm * 16, l16, lh);
        }
#pragma unroll
        for (int tn = 0; tn < 2; ++tn) {
            int cb = wn * 32 + tn * 16;
            v16bf bH = load_b_frag(sBT_hi, cb, l16, lh);
            v16bf bL = load_b_frag(sBT_lo, cb, l16, lh);
#pragma unroll
            for (int tm = 0; tm < 2; ++tm) {
                acc[tm][tn] = wmma_bf16(aL[tm], bH, acc[tm][tn]);
                acc[tm][tn] = wmma_bf16(aH[tm], bL, acc[tm][tn]);
                acc[tm][tn] = wmma_bf16(aH[tm], bH, acc[tm][tn]);
            }
        }
        __syncthreads();
    }

    // ---- epilogue: gate scale + accumulate into y ----
#pragma unroll
    for (int tm = 0; tm < 2; ++tm)
#pragma unroll
        for (int tn = 0; tn < 2; ++tn)
#pragma unroll
            for (int r = 0; r < 8; ++r) {
                int rloc = wm * 32 + tm * 16 + lh * 8 + r;
                int col  = colBlk + wn * 32 + tn * 16 + l16;
                float v = sG[rloc] * acc[tm][tn][r];
                size_t o = (size_t)(rowBlk + rloc) * D_MODEL + col;
                if (beta) v += y[o];
                y[o] = v;
            }
}

// =====================================================================
extern "C" void kernel_launch(void* const* d_in, const int* in_sizes, int n_in,
                              void* d_out, int out_size, void* d_ws, size_t ws_size,
                              hipStream_t stream) {
    (void)in_sizes; (void)n_in; (void)out_size; (void)ws_size;

    const float* x           = (const float*)d_in[0];
    const float* expert_p    = (const float*)d_in[1];
    const int*   expert_idxs = (const int*)d_in[2];
    const float* w_up        = (const float*)d_in[3];
    const float* w_down      = (const float*)d_in[4];
    float*       y           = (float*)d_out;

    float* G   = (float*)d_ws;                                                // [N, E]
    float* act = (float*)((char*)d_ws + (size_t)N_TOKENS * NUM_EXPERTS * sizeof(float)); // [N, H]

    moe_gates_kernel<<<dim3(N_TOKENS / 256), dim3(256), 0, stream>>>(expert_p, expert_idxs, G);

    for (int e = 0; e < NUM_EXPERTS; ++e) {
        moe_up_kernel<<<dim3(HIDDEN / BN, N_TOKENS / BM), dim3(256), 0, stream>>>(
            x, w_up + (size_t)e * D_MODEL * TWO_H, act);
        moe_down_kernel<<<dim3(D_MODEL / BN, N_TOKENS / BM), dim3(256), 0, stream>>>(
            act, w_down + (size_t)e * HIDDEN * D_MODEL, G, e, (e == 0) ? 0 : 1, y);
    }
}
